// GRUPK_83588653515347
// MI455X (gfx1250) — matrix operand order repaired and checked by reading, not compiled, not run on hardware
//
#include <hip/hip_runtime.h>

// ---------------------------------------------------------------------------
// GRU (2-layer, B=256, S=1024, H=256, G=768) fused persistent kernel, gfx1250.
// bf16 WMMA (v_wmma_f32_16x16x32_bf16) for all matmuls, fp32 accumulate,
// fp32 master hidden state in registers (WMMA C-layout), bf16 hidden in LDS.
// 16 waves/WG, each wave owns a 16-wide hidden slice (r/z/n tiles together),
// keeping accumulators+fragments under 256 VGPRs (no MSB swaps, no spills).
// An opaque integer offset (laundered per step) blocks LICM from hoisting the
// loop-invariant weight fragments while keeping loads on the global path.
// ---------------------------------------------------------------------------

typedef float  floatx8 __attribute__((ext_vector_type(8)));
typedef __bf16 bf16x8  __attribute__((ext_vector_type(8)));
typedef __bf16 bf16x16 __attribute__((ext_vector_type(16)));

#define S_SZ 1024
#define H_SZ 256
#define G_SZ 768
#define BLOCK 512           // 16 waves

// workspace layout (bf16 element offsets)
#define WS_WHH0  0          // 768 x 256
#define WS_WIH1  196608     // 768 x 256
#define WS_WHH1  393216     // 768 x 256
#define WS_WIH0P 589824     // 768 x 32 (W_ih0 zero-padded K: 8 -> 32)
#define WS_TOTAL 614400

#define HST 264             // LDS row stride (halfwords) for hidden buffers
#define XST 40              // LDS row stride for padded input tile (16x32 used)

// -------------------------- prologue: fp32 -> bf16 -------------------------
__global__ void grupk_cvt_kernel(const float* __restrict__ Whh0,
                                 const float* __restrict__ Wih1,
                                 const float* __restrict__ Whh1,
                                 const float* __restrict__ Wih0,
                                 __bf16* __restrict__ wsb) {
  int stride = gridDim.x * blockDim.x;
  for (int i = blockIdx.x * blockDim.x + threadIdx.x; i < WS_TOTAL; i += stride) {
    float v;
    if (i < WS_WIH1)        v = Whh0[i];
    else if (i < WS_WHH1)   v = Wih1[i - WS_WIH1];
    else if (i < WS_WIH0P)  v = Whh1[i - WS_WHH1];
    else { int j = i - WS_WIH0P; int g = j >> 5, k = j & 31;
           v = (k < 8) ? Wih0[g * 8 + k] : 0.f; }
    wsb[i] = (__bf16)v;
  }
}

// ----------------------------- fragment loads ------------------------------
// A (16x32 bf16, M x K): lane l holds row M=l&15; 16 consecutive K values
// starting at (l>>4)*16 -> two contiguous 16-byte LDS loads.
__device__ __forceinline__ bf16x16 load_a(const __bf16* base, int stride,
                                          int kt, int lane) {
  const __bf16* p = base + (lane & 15) * stride + kt * 32 + ((lane >> 4) * 16);
  bf16x8 lo = *(const bf16x8*)p;
  bf16x8 hi = *(const bf16x8*)(p + 8);
  return __builtin_shufflevector(lo, hi, 0,1,2,3,4,5,6,7,8,9,10,11,12,13,14,15);
}

// B (32x16 bf16, K x N): lane l holds column N=nbase+(l&15), 16 consecutive
// K values starting at (l>>4)*16 -> 32 contiguous bytes of the row-major
// (G x K) weight matrix. Two global_load_b128 per lane (L2 resident).
__device__ __forceinline__ bf16x16 load_b(const __bf16* W, int stride,
                                          int nbase, int kt, int lane) {
  const __bf16* p = W + (size_t)(nbase + (lane & 15)) * stride
                      + kt * 32 + ((lane >> 4) * 16);
  bf16x8 lo = *(const bf16x8*)p;
  bf16x8 hi = *(const bf16x8*)(p + 8);
  return __builtin_shufflevector(lo, hi, 0,1,2,3,4,5,6,7,8,9,10,11,12,13,14,15);
}

__device__ __forceinline__ floatx8 wmma_acc(bf16x16 a, bf16x16 b, floatx8 c) {
  return __builtin_amdgcn_wmma_f32_16x16x32_bf16(
      /*neg_a=*/false, a, /*neg_b=*/false, b,
      /*c_mod=*/(short)0, c, /*reuse_a=*/false, /*reuse_b=*/false);
}

template <int KT>
__device__ __forceinline__ void mm3(floatx8 acc[3], const __bf16* A, int astride,
                                    const __bf16* B, int bstride,
                                    const int nb[3], int lane) {
#pragma unroll
  for (int kt = 0; kt < KT; kt++) {
    bf16x16 a = load_a(A, astride, kt, lane);
#pragma unroll
    for (int i = 0; i < 3; i++) {
      bf16x16 b = load_b(B, bstride, nb[i], kt, lane);
      acc[i] = wmma_acc(a, b, acc[i]);
    }
  }
}

__device__ __forceinline__ float sigm(float x) {
  return 1.f / (1.f + __expf(-x));
}
__device__ __forceinline__ float tanh_fast(float x) {
  float e = __expf(-2.f * x);
  return (1.f - e) / (1.f + e);
}
__device__ __forceinline__ floatx8 splat8(float v) {
  floatx8 r = {v, v, v, v, v, v, v, v};
  return r;
}

// r/z/n combine + h update in WMMA C-layout registers; scatter bf16 h to LDS.
__device__ __forceinline__ void gate_update(const floatx8 ax[3], const floatx8 ah[3],
                                            floatx8& hreg, __bf16* hlds,
                                            int lane, int jlo) {
  const int j = jlo + (lane & 15);
#pragma unroll
  for (int v = 0; v < 8; v++) {
    float r = sigm(ax[0][v] + ah[0][v]);
    float z = sigm(ax[1][v] + ah[1][v]);
    float n = tanh_fast(ax[2][v] + r * ah[2][v]);
    float h = (1.f - z) * n + z * hreg[v];
    hreg[v] = h;
    int M = v + ((lane >> 4) << 3);
    hlds[M * HST + j] = (__bf16)h;
  }
}

// ------------------------------ main kernel --------------------------------
__global__ __launch_bounds__(BLOCK, 1)
__attribute__((amdgpu_waves_per_eu(4)))
void grupk_main_kernel(const float* __restrict__ x,
                       const float* __restrict__ meta,
                       const float* __restrict__ b_ih0,
                       const float* __restrict__ b_hh0,
                       const float* __restrict__ b_ih1,
                       const float* __restrict__ b_hh1,
                       const float* __restrict__ fcW,
                       const float* __restrict__ fcb,
                       const __bf16* __restrict__ wsb,
                       float* __restrict__ out) {
  __shared__ __bf16 xinb[16 * XST];   // concat(x_t, meta) tile, zero padded K
  __shared__ __bf16 h1b[16 * HST];    // layer-0 hidden, bf16
  __shared__ __bf16 h2b[16 * HST];    // layer-1 hidden, bf16
  __shared__ float  hf[16 * H_SZ];    // final h2 fp32 for FC
  __shared__ float  red[16 * 17];

  const int tid  = threadIdx.x;
  const int lane = tid & 31;
  const int w    = tid >> 5;          // 16 waves
  const int bb   = blockIdx.x * 16;   // batch base for this WG
  const int jlo  = w * 16;            // 16-wide hidden slice owned by this wave

  // ---- init LDS state
  for (int i = tid; i < 16 * XST; i += BLOCK) xinb[i] = (__bf16)0.f;
  for (int i = tid; i < 16 * HST; i += BLOCK) { h1b[i] = (__bf16)0.f; h2b[i] = (__bf16)0.f; }
  __syncthreads();
  if (tid < 16) {
#pragma unroll
    for (int f = 0; f < 4; f++)
      xinb[tid * XST + 4 + f] = (__bf16)meta[(size_t)(bb + tid) * 4 + f];
  }

  // ---- per-lane gate-column map + bias values (N = column, same for all vgprs)
  int nb[3];
  float bih0v[3], bhh0v[3], bih1v[3], bhh1v[3];
#pragma unroll
  for (int i = 0; i < 3; i++) {
    nb[i] = i * H_SZ + jlo;           // r/z/n gate sections, this wave's slice
    int g = nb[i] + (lane & 15);
    bih0v[i] = b_ih0[g]; bhh0v[i] = b_hh0[g];
    bih1v[i] = b_ih1[g]; bhh1v[i] = b_hh1[g];
  }

  floatx8 h1r = splat8(0.f);
  floatx8 h2r = splat8(0.f);
  __syncthreads();

  // ---- recurrence over time
#pragma clang loop unroll(disable)
  for (int t = 0; t < S_SZ; t++) {
    // Opaque zero offset, refreshed each step: keeps weight-fragment loads
    // inside the loop (no LICM hoist -> no spill) while preserving the
    // global-pointer provenance of wsb (global_load_b128, not flat).
    int zoff = 0;
    asm volatile("" : "+s"(zoff));
    const __bf16* Whh0b = wsb + (WS_WHH0  + zoff);
    const __bf16* Wih1b = wsb + (WS_WIH1  + zoff);
    const __bf16* Whh1b = wsb + (WS_WHH1  + zoff);
    const __bf16* Wih0p = wsb + (WS_WIH0P + zoff);

    if (tid < 16) {
      const float4 xv = *(const float4*)(x + ((size_t)(bb + tid) * S_SZ + t) * 4);
      xinb[tid * XST + 0] = (__bf16)xv.x;
      xinb[tid * XST + 1] = (__bf16)xv.y;
      xinb[tid * XST + 2] = (__bf16)xv.z;
      xinb[tid * XST + 3] = (__bf16)xv.w;
    }
    __syncthreads();                                   // S1: x_t staged

    // ----- layer 0: xg = [x_t,meta] @ Wih0p^T + b ; hg = h1 @ Whh0^T + b
    floatx8 ax[3], ah[3];
#pragma unroll
    for (int i = 0; i < 3; i++) { ax[i] = splat8(bih0v[i]); ah[i] = splat8(bhh0v[i]); }
    mm3<1>(ax, xinb, XST, Wih0p, 32,   nb, lane);
    mm3<8>(ah, h1b,  HST, Whh0b, H_SZ, nb, lane);
    __syncthreads();                                   // S2: h1b reads done
    gate_update(ax, ah, h1r, h1b, lane, jlo);          // writes new h1 bf16
    __syncthreads();                                   // S3: h1b ready

    // ----- layer 1: xg = h1 @ Wih1^T + b ; hg = h2 @ Whh1^T + b
#pragma unroll
    for (int i = 0; i < 3; i++) { ax[i] = splat8(bih1v[i]); ah[i] = splat8(bhh1v[i]); }
    mm3<8>(ax, h1b, HST, Wih1b, H_SZ, nb, lane);
    mm3<8>(ah, h2b, HST, Whh1b, H_SZ, nb, lane);
    __syncthreads();                                   // S4: h2b reads done
    gate_update(ax, ah, h2r, h2b, lane, jlo);          // writes new h2 bf16
    // next-iteration S1 orders these stores vs. next layer-1 reads
  }

  // ---- epilogue: out[b] = h2[b,:] . fcW + fcb
#pragma unroll
  for (int v = 0; v < 8; v++) {
    int M = v + ((lane >> 4) << 3);
    int j = jlo + (lane & 15);
    hf[M * H_SZ + j] = h2r[v];
  }
  __syncthreads();
  if (tid < 256) {
    int row = tid >> 4, seg = tid & 15;
    float p = 0.f;
#pragma unroll
    for (int c = 0; c < 16; c++)
      p += hf[row * H_SZ + seg * 16 + c] * fcW[seg * 16 + c];
    red[row * 17 + seg] = p;
  }
  __syncthreads();
  if (tid < 256 && (tid & 15) == 0) {
    int row = tid >> 4;
    float s = fcb[0];
#pragma unroll
    for (int k2 = 0; k2 < 16; k2++) s += red[row * 17 + k2];
    out[bb + row] = s;
  }
}

// ------------------------------- launcher ----------------------------------
extern "C" void kernel_launch(void* const* d_in, const int* in_sizes, int n_in,
                              void* d_out, int out_size, void* d_ws, size_t ws_size,
                              hipStream_t stream) {
  (void)in_sizes; (void)n_in; (void)out_size; (void)ws_size;
  const float* x     = (const float*)d_in[0];
  const float* meta  = (const float*)d_in[1];
  const float* W_ih0 = (const float*)d_in[2];
  const float* W_hh0 = (const float*)d_in[3];
  const float* b_ih0 = (const float*)d_in[4];
  const float* b_hh0 = (const float*)d_in[5];
  const float* W_ih1 = (const float*)d_in[6];
  const float* W_hh1 = (const float*)d_in[7];
  const float* b_ih1 = (const float*)d_in[8];
  const float* b_hh1 = (const float*)d_in[9];
  const float* fc_W  = (const float*)d_in[10];
  const float* fc_b  = (const float*)d_in[11];
  float* out = (float*)d_out;
  __bf16* wsb = (__bf16*)d_ws;

  grupk_cvt_kernel<<<600, 256, 0, stream>>>(W_hh0, W_ih1, W_hh1, W_ih0, wsb);
  grupk_main_kernel<<<16, BLOCK, 0, stream>>>(x, meta, b_ih0, b_hh0, b_ih1, b_hh1,
                                              fc_W, fc_b, wsb, out);
}
